// Gnn_basic_9715216023966
// MI455X (gfx1250) — compile-verified, matrix operand order checked by reading, not run on hardware
//
#include <hip/hip_runtime.h>

typedef _Float16 f16;
typedef __attribute__((ext_vector_type(16))) _Float16 v16h;
typedef __attribute__((ext_vector_type(8)))  _Float16 v8h;
typedef __attribute__((ext_vector_type(8)))  float    v8f;

constexpr int Bv = 2, Nn = 50000, Ee = 800000;

// Packed-weight offsets (in halfs). Layout per matrix:
//   [nTiles][nChunks][lane(32)][16 halfs]  -> one v16h B-operand per lane.
enum : int {
  OFF_EE1 = 0,      OFF_EE2 = 2048,        // ee: K=8(pad32) N=64 ; K=64 N=64
  OFF_NE1 = 6144,   OFF_NE2 = 8192,        // ne: K=16(pad32)     ; K=64
  OFF_EP1 = 12288,  OFF_EP2 = 24576,       // ep: K=192           ; K=64
  OFF_NPR1 = 28672, OFF_NPR2 = 36864,      // npr: K=128          ; K=64
  OFF_DEC1 = 40960, OFF_DEC2 = 45056,      // dec: K=64 N=64      ; K=64 N=16
  WP_HALFS = 46080
};

__device__ __forceinline__ v16h hcat(v8h lo, v8h hi) {
  return __builtin_shufflevector(lo, hi, 0,1,2,3,4,5,6,7,8,9,10,11,12,13,14,15);
}

// One K=32 WMMA step over NT 16-col output tiles.
template<int NT>
__device__ __forceinline__ void wmma_step(v8f* acc, v16h a, const f16* wpk,
                                          int chunk, int nChunks, int lane) {
#pragma unroll
  for (int t = 0; t < NT; ++t) {
    const v16h b = *(const v16h*)(wpk + ((size_t)(t * nChunks + chunk) * 32 + lane) * 16);
    acc[t] = __builtin_amdgcn_wmma_f32_16x16x32_f16(
        false, a, false, b, (short)0, acc[t], false, false);
  }
}

// C/D-layout accumulators (+bias, optional ReLU) -> 16x64 f16 tile in LDS.
// Tile is private to the wave; per-wave DS ops are in order, the asm clobber
// keeps the compiler from reordering across it.
__device__ __forceinline__ void tile_to_lds(f16* T, const v8f* acc,
                                            const float* __restrict__ bias,
                                            int lane, bool relu) {
  const int n = lane & 15, half = lane >> 4;
#pragma unroll
  for (int t = 0; t < 4; ++t) {
    const float bb = bias[t * 16 + n];
#pragma unroll
    for (int j = 0; j < 8; ++j) {
      float v = acc[t][j] + bb;
      if (relu) v = v > 0.f ? v : 0.f;
      T[(j + 8 * half) * 72 + t * 16 + n] = (f16)v;
    }
  }
  asm volatile("" ::: "memory");
}

// Read A-operand (row m = lane&15, K-chunk `chunk`) from a 16x64 LDS tile.
__device__ __forceinline__ v16h lds_a(const f16* T, int lane, int chunk) {
  const int m = lane & 15, kb = (lane >> 4) * 8;
  const f16* row = T + m * 72 + chunk * 32;
  v8h lo = *(const v8h*)(row + kb);
  v8h hi = *(const v8h*)(row + 16 + kb);
  return hcat(lo, hi);
}

// Copy a finished 16x64 tile from LDS to global rows (stride 64 halfs).
__device__ __forceinline__ void lds_rows_to_global(const f16* T, f16* dst, int lane) {
  const int m = lane & 15, part = lane >> 4;
  const f16* s = T + m * 72 + part * 32;
  f16* d = dst + (size_t)m * 64 + part * 32;
#pragma unroll
  for (int i = 0; i < 4; ++i) ((v8h*)d)[i] = *(const v8h*)(s + i * 8);
}

// Cooperative block copy of packed weights into LDS, then barrier.
__device__ __forceinline__ void stage_weights(f16* dst, const f16* src, int halfs) {
  const int vecs = halfs >> 3;
  for (int i = threadIdx.x; i < vecs; i += blockDim.x)
    ((v8h*)dst)[i] = ((const v8h*)src)[i];
  __syncthreads();
}

// ---------------- weight packing ----------------
struct WPtrs { const float* W[10]; };

__global__ void __launch_bounds__(256) k_pack(WPtrs p, f16* __restrict__ wp) {
  const int K [10] = {8, 64, 16, 64, 192, 64, 128, 64, 64, 64};
  const int Nf[10] = {64, 64, 64, 64, 64, 64, 64, 64, 64, 16};
  const int nC[10] = {1, 2, 1, 2, 6, 2, 4, 2, 2, 2};
  const int nT[10] = {4, 4, 4, 4, 4, 4, 4, 4, 4, 1};
  const int off[10] = {OFF_EE1, OFF_EE2, OFF_NE1, OFF_NE2, OFF_EP1,
                       OFF_EP2, OFF_NPR1, OFF_NPR2, OFF_DEC1, OFF_DEC2};
  const int m = blockIdx.y;
  const int idx = blockIdx.x * blockDim.x + threadIdx.x;
  const int total = nT[m] * nC[m] * 512;
  if (idx >= total) return;
  const int e = idx & 15;
  const int lane = (idx >> 4) & 31;
  const int ct = idx >> 9;
  const int chunk = ct % nC[m];
  const int n = (ct / nC[m]) * 16 + (lane & 15);
  const int kb = (lane >> 4) * 8;
  const int i = e >> 1, pb = e & 1;
  const int kin = (i < 4) ? (kb + 2 * i + pb) : (16 + kb + 2 * (i - 4) + pb);
  const int k = chunk * 32 + kin;
  const float v = (k < K[m]) ? p.W[m][k * Nf[m] + n] : 0.f;
  wp[off[m] + idx] = (f16)v;
}

// ---------------- node encoder: hn = mlp_ne(nf) ----------------
__global__ void __launch_bounds__(256) k_node_enc(
    const float* __restrict__ nf, const f16* __restrict__ wp,
    const float* __restrict__ b1, const float* __restrict__ b2,
    f16* __restrict__ hn) {
  __shared__ f16 wsh[6144];
  __shared__ f16 tls[8 * 16 * 72];
  stage_weights(wsh, wp + OFF_NE1, 6144);            // NE1 @0, NE2 @2048
  const int wave = threadIdx.x >> 5, lane = threadIdx.x & 31;
  const int gw = blockIdx.x * 8 + wave;
  const int tilesB = Nn / 16;
  if (gw >= Bv * tilesB) return;
  const int b = gw / tilesB, tile = gw % tilesB;
  f16* T = tls + wave * 16 * 72;
  const int m = lane & 15, kb = (lane >> 4) * 8;
  const size_t row0 = (size_t)b * Nn + (size_t)tile * 16;

  // layer 1: K=16 padded to 32 (run1 = zeros, weights zero-padded too)
  const float* s0 = nf + (row0 + m) * 16 + kb;
  v8h lo;
#pragma unroll
  for (int i = 0; i < 8; ++i) lo[i] = (f16)s0[i];
  v8h hi = {};
  v8f acc[4] = {};
  wmma_step<4>(acc, hcat(lo, hi), wsh + 0, 0, 1, lane);
  tile_to_lds(T, acc, b1, lane, true);

  // layer 2: K=64
  v8f acc2[4] = {};
#pragma unroll
  for (int c = 0; c < 2; ++c)
    wmma_step<4>(acc2, lds_a(T, lane, c), wsh + 2048, c, 2, lane);
  tile_to_lds(T, acc2, b2, lane, false);
  lds_rows_to_global(T, hn + row0 * 64, lane);
}

// ---------------- edge encoder: he = mlp_ee(ef) ----------------
__global__ void __launch_bounds__(256) k_edge_enc(
    const float* __restrict__ ef, const f16* __restrict__ wp,
    const float* __restrict__ b1, const float* __restrict__ b2,
    f16* __restrict__ he) {
  __shared__ f16 wsh[6144];
  __shared__ f16 tls[8 * 16 * 72];
  stage_weights(wsh, wp + OFF_EE1, 6144);            // EE1 @0, EE2 @2048
  const int wave = threadIdx.x >> 5, lane = threadIdx.x & 31;
  const int gw = blockIdx.x * 8 + wave;
  const int tilesB = Ee / 16;
  if (gw >= Bv * tilesB) return;
  const int b = gw / tilesB, tile = gw % tilesB;
  f16* T = tls + wave * 16 * 72;
  const int m = lane & 15, half = lane >> 4;
  const size_t row0 = (size_t)b * Ee + (size_t)tile * 16;

  // layer 1: K=8 padded to 32 (only lanes 0-15 carry data: k in [0,8))
  v8h lo = {};
  if (half == 0) {
    const float* s0 = ef + (row0 + m) * 8;
#pragma unroll
    for (int i = 0; i < 8; ++i) lo[i] = (f16)s0[i];
  }
  v8h hi = {};
  v8f acc[4] = {};
  wmma_step<4>(acc, hcat(lo, hi), wsh + 0, 0, 1, lane);
  tile_to_lds(T, acc, b1, lane, true);

  v8f acc2[4] = {};
#pragma unroll
  for (int c = 0; c < 2; ++c)
    wmma_step<4>(acc2, lds_a(T, lane, c), wsh + 2048, c, 2, lane);
  tile_to_lds(T, acc2, b2, lane, false);
  lds_rows_to_global(T, he + row0 * 64, lane);
}

// ------- edge processor: he2 = mlp_ep([he, hn[s], hn[r]]); agg += scatter -------
__global__ void __launch_bounds__(256) k_edge_proc(
    const int* __restrict__ edges, const f16* __restrict__ he,
    const f16* __restrict__ hn, const f16* __restrict__ wp,
    const float* __restrict__ b1, const float* __restrict__ b2,
    float* __restrict__ agg) {
  __shared__ f16 wsh[16384];
  __shared__ f16 tls[8 * 16 * 72];
  stage_weights(wsh, wp + OFF_EP1, 16384);           // EP1 @0, EP2 @12288
  const int wave = threadIdx.x >> 5, lane = threadIdx.x & 31;
  const int gw = blockIdx.x * 8 + wave;
  const int tilesB = Ee / 16;
  if (gw >= Bv * tilesB) return;
  const int b = gw / tilesB, tile = gw % tilesB;
  f16* T = tls + wave * 16 * 72;
  const int m = lane & 15, half = lane >> 4, kb = half * 8;
  const int* e0 = edges + (size_t)b * 2 * Ee;
  const int* e1 = e0 + Ee;
  const size_t edge = (size_t)tile * 16 + m;
  const int s = e0[edge], r = e1[edge];
  const f16* heRow = he + ((size_t)b * Ee + edge) * 64;
  const f16* hsRow = hn + ((size_t)b * Nn + s) * 64;
  const f16* hrRow = hn + ((size_t)b * Nn + r) * 64;

  // layer 1: K=192 = [he(64) | hn[s](64) | hn[r](64)], gathered on the fly
  v8f acc[4] = {};
#pragma unroll
  for (int c = 0; c < 6; ++c) {
    const f16* src = (c < 2) ? heRow : ((c < 4) ? hsRow : hrRow);
    const int off = (c & 1) * 32;
    v8h lo = *(const v8h*)(src + off + kb);
    v8h hi = *(const v8h*)(src + off + 16 + kb);
    wmma_step<4>(acc, hcat(lo, hi), wsh + 0, c, 6, lane);
  }
  tile_to_lds(T, acc, b1, lane, true);

  // layer 2: K=64
  v8f acc2[4] = {};
#pragma unroll
  for (int c = 0; c < 2; ++c)
    wmma_step<4>(acc2, lds_a(T, lane, c), wsh + 12288, c, 2, lane);

  // segment_sum scatter: agg[b, e0[edge], :] += out  (f32 atomics into L2)
  const int n = lane & 15;
  int seg[8];
#pragma unroll
  for (int j = 0; j < 8; ++j) seg[j] = e0[(size_t)tile * 16 + j + 8 * half];
#pragma unroll
  for (int t = 0; t < 4; ++t) {
    const float bb = b2[t * 16 + n];
#pragma unroll
    for (int j = 0; j < 8; ++j)
      atomicAdd(agg + ((size_t)b * Nn + seg[j]) * 64 + t * 16 + n, acc2[t][j] + bb);
  }
}

// ------- node processor + decoder + residual -------
__global__ void __launch_bounds__(256) k_node_dec(
    const float* __restrict__ nf, const f16* __restrict__ hn,
    const float* __restrict__ agg, const f16* __restrict__ wp,
    const float* __restrict__ nb1, const float* __restrict__ nb2,
    const float* __restrict__ db1, const float* __restrict__ db2,
    float* __restrict__ out) {
  __shared__ f16 wsh[17408];
  __shared__ f16 tls[8 * 16 * 72];
  stage_weights(wsh, wp + OFF_NPR1, 17408);  // NPR1@0 NPR2@8192 DEC1@12288 DEC2@16384
  const int wave = threadIdx.x >> 5, lane = threadIdx.x & 31;
  const int gw = blockIdx.x * 8 + wave;
  const int tilesB = Nn / 16;
  if (gw >= Bv * tilesB) return;
  const int b = gw / tilesB, tile = gw % tilesB;
  f16* T = tls + wave * 16 * 72;
  const int m = lane & 15, half = lane >> 4, kb = half * 8;
  const size_t row0 = (size_t)b * Nn + (size_t)tile * 16;
  const f16* hnRow = hn + (row0 + m) * 64;
  const float* agRow = agg + (row0 + m) * 64;

  // npr layer 1: K=128 = [hn(64,f16) | agg(64,f32->f16)]
  v8f acc[4] = {};
#pragma unroll
  for (int c = 0; c < 4; ++c) {
    v8h lo, hi;
    if (c < 2) {
      lo = *(const v8h*)(hnRow + c * 32 + kb);
      hi = *(const v8h*)(hnRow + c * 32 + 16 + kb);
    } else {
      const float* a0 = agRow + (c - 2) * 32 + kb;
#pragma unroll
      for (int i = 0; i < 8; ++i) { lo[i] = (f16)a0[i]; hi[i] = (f16)a0[16 + i]; }
    }
    wmma_step<4>(acc, hcat(lo, hi), wsh + 0, c, 4, lane);
  }
  tile_to_lds(T, acc, nb1, lane, true);

  // npr layer 2 -> hn2
  v8f acc2[4] = {};
#pragma unroll
  for (int c = 0; c < 2; ++c)
    wmma_step<4>(acc2, lds_a(T, lane, c), wsh + 8192, c, 2, lane);
  tile_to_lds(T, acc2, nb2, lane, false);

  // dec layer 1
  v8f acc3[4] = {};
#pragma unroll
  for (int c = 0; c < 2; ++c)
    wmma_step<4>(acc3, lds_a(T, lane, c), wsh + 12288, c, 2, lane);
  tile_to_lds(T, acc3, db1, lane, true);

  // dec layer 2: N=16 (single tile) + residual
  v8f acc4 = {};
#pragma unroll
  for (int c = 0; c < 2; ++c)
    wmma_step<1>(&acc4, lds_a(T, lane, c), wsh + 16384, c, 2, lane);
  const int n = lane & 15;
  const float bb = db2[n];
#pragma unroll
  for (int j = 0; j < 8; ++j) {
    const size_t o = (row0 + j + 8 * half) * 16 + n;
    out[o] = nf[o] + acc4[j] + bb;
  }
}

extern "C" void kernel_launch(void* const* d_in, const int* in_sizes, int n_in,
                              void* d_out, int out_size, void* d_ws, size_t ws_size,
                              hipStream_t stream) {
  (void)in_sizes; (void)n_in; (void)out_size; (void)ws_size;
  const int*   edges = (const int*)  d_in[0];
  const float* nf    = (const float*)d_in[1];
  const float* ef    = (const float*)d_in[2];
  WPtrs wpt;
  const int widx[10] = {3, 5, 7, 9, 11, 13, 15, 17, 19, 21};
  for (int i = 0; i < 10; ++i) wpt.W[i] = (const float*)d_in[widx[i]];
  const float* eeb1 = (const float*)d_in[4];
  const float* eeb2 = (const float*)d_in[6];
  const float* neb1 = (const float*)d_in[8];
  const float* neb2 = (const float*)d_in[10];
  const float* epb1 = (const float*)d_in[12];
  const float* epb2 = (const float*)d_in[14];
  const float* nprb1 = (const float*)d_in[16];
  const float* nprb2 = (const float*)d_in[18];
  const float* decb1 = (const float*)d_in[20];
  const float* decb2 = (const float*)d_in[22];

  char* ws   = (char*)d_ws;
  f16*   wp  = (f16*)ws;                                           // 92160 B
  f16*   hn  = (f16*)(ws + (size_t)WP_HALFS * 2);                  // B*N*64 f16
  f16*   he  = (f16*)((char*)hn + (size_t)Bv * Nn * 64 * 2);       // B*E*64 f16
  float* agg = (float*)((char*)he + (size_t)Bv * Ee * 64 * 2);     // B*N*64 f32

  k_pack<<<dim3(48, 10), 256, 0, stream>>>(wpt, wp);
  hipMemsetAsync(agg, 0, (size_t)Bv * Nn * 64 * sizeof(float), stream);

  const int nodeBlocks = (Bv * (Nn / 16) + 7) / 8;   // 782
  const int edgeBlocks = (Bv * (Ee / 16) + 7) / 8;   // 12500
  k_node_enc<<<nodeBlocks, 256, 0, stream>>>(nf, wp, neb1, neb2, hn);
  k_edge_enc<<<edgeBlocks, 256, 0, stream>>>(ef, wp, eeb1, eeb2, he);
  k_edge_proc<<<edgeBlocks, 256, 0, stream>>>(edges, he, hn, wp, epb1, epb2, agg);
  k_node_dec<<<nodeBlocks, 256, 0, stream>>>(nf, hn, agg, wp,
                                             nprb1, nprb2, decb1, decb2,
                                             (float*)d_out);
}
